// Model_52639119180522
// MI455X (gfx1250) — matrix operand order repaired and checked by reading, not compiled, not run on hardware
//
#include <hip/hip_runtime.h>
#include <hip/hip_bf16.h>
#include <math.h>

typedef __attribute__((ext_vector_type(16))) __bf16       v16bf;
typedef __attribute__((ext_vector_type(8)))  float        v8f;
typedef __attribute__((ext_vector_type(4)))  unsigned int u32x4;

union BF16Tile {
    v16bf v;
    u32x4 q[2];
    unsigned short u[16];
};

#define BB   32
#define SS   512
#define WINW 5
#define EMBD 300
#define K1   1500
#define K1P  1504
#define HH   300
#define HP   320
#define NF   128

__device__ __forceinline__ unsigned short f2bf(float f) {
    unsigned int u = __float_as_uint(f);
    return (unsigned short)((u + 0x7FFFu + ((u >> 16) & 1u)) >> 16);  // RNE
}

// ---- pad + convert f32 weight [N][K] -> bf16 [Npad][Kpad], zero pad ----
__global__ void padcvt_kernel(const float* __restrict__ src, unsigned short* __restrict__ dst,
                              int N, int K, int Npad, int Kpad) {
    int idx = blockIdx.x * blockDim.x + threadIdx.x;
    if (idx >= Npad * Kpad) return;
    int n = idx / Kpad, k = idx - n * Kpad;
    unsigned short v = 0;
    if (n < N && k < K) v = f2bf(src[(size_t)n * K + k]);
    dst[idx] = v;
}

// ---- combined bias (b_ih + b_hh), zero padded ----
__global__ void bias_kernel(const float* __restrict__ a, const float* __restrict__ b,
                            float* __restrict__ dst, int N, int Npad) {
    int i = blockIdx.x * blockDim.x + threadIdx.x;
    if (i >= Npad) return;
    dst[i] = (i < N) ? (a[i] + b[i]) : 0.f;
}

// ---- embedding gather + relu + bf16, seq-major rows m = s*B + b, K padded ----
__global__ void embed_kernel(const int* __restrict__ x, const float* __restrict__ emb,
                             unsigned short* __restrict__ XE) {
    int idx = blockIdx.x * blockDim.x + threadIdx.x;
    if (idx >= BB * SS * K1P) return;
    int m  = idx / K1P;
    int kp = idx - m * K1P;
    unsigned short v = 0;
    if (kp < K1) {
        int s = m >> 5, b = m & 31;
        int w = kp / EMBD, e = kp - w * EMBD;
        int tok = x[(b * SS + s) * WINW + w];
        float f = emb[(size_t)tok * EMBD + e];
        v = f2bf(f > 0.f ? f : 0.f);
    }
    XE[idx] = v;
}

// ---- helpers for WMMA operand fetch (layouts per CDNA5 ISA tables) ----
__device__ __forceinline__ void loadA(BF16Tile& a, const unsigned short* row, int kt, int khalf) {
    a.q[0] = *(const u32x4*)(row + kt + khalf * 8);
    a.q[1] = *(const u32x4*)(row + kt + 16 + khalf * 8);
}
__device__ __forceinline__ void loadB(BF16Tile& b, const unsigned short* col, int kt, int khalf) {
    b.q[0] = *(const u32x4*)(col + kt + khalf * 8);
    b.q[1] = *(const u32x4*)(col + kt + 16 + khalf * 8);
}
__device__ __forceinline__ v8f wmma_bf16(const BF16Tile& a, const BF16Tile& b, v8f c) {
    return __builtin_amdgcn_wmma_f32_16x16x32_bf16(false, a.v, false, b.v,
                                                   (short)0, c, false, false);
}

// ---- bf16 WMMA GEMM, 2x2 register blocking (32x32 macro-tile per wave):
//      C[M][Npad] = A[M][ldk] @ W[Npad][ldk]^T + bias ----
__global__ void __launch_bounds__(256) wmma_gemm_kernel(
    const unsigned short* __restrict__ A, const unsigned short* __restrict__ W,
    const float* __restrict__ bias, float* __restrict__ C,
    int M, int Npad, int ldk) {
    int gw   = (int)((blockIdx.x * blockDim.x + threadIdx.x) >> 5);
    int lane = threadIdx.x & 31;
    int tilesN = Npad >> 5;                    // 32-wide macro tiles in N
    int mo = (gw / tilesN) << 5;
    int no = (gw - (gw / tilesN) * tilesN) << 5;
    if (mo >= M) return;                       // wave-uniform, EXEC stays all-1
    int khalf = lane >> 4, l15 = lane & 15;

    float bv0 = bias[no + l15];
    float bv1 = bias[no + 16 + l15];
    v8f acc00, acc01, acc10, acc11;
#pragma unroll
    for (int v = 0; v < 8; ++v) { acc00[v] = bv0; acc01[v] = bv1; acc10[v] = bv0; acc11[v] = bv1; }

    const unsigned short* Arow0 = A + (size_t)(mo + l15) * ldk;
    const unsigned short* Arow1 = A + (size_t)(mo + 16 + l15) * ldk;
    const unsigned short* Wcol0 = W + (size_t)(no + l15) * ldk;
    const unsigned short* Wcol1 = W + (size_t)(no + 16 + l15) * ldk;

    for (int kt = 0; kt < ldk; kt += 32) {
        __builtin_prefetch(Arow0 + kt + 512, 0, 1);   // global_prefetch_b8, next K chunk
        BF16Tile a0, a1, b0, b1;
        loadA(a0, Arow0, kt, khalf);
        loadA(a1, Arow1, kt, khalf);
        loadB(b0, Wcol0, kt, khalf);
        loadB(b1, Wcol1, kt, khalf);
        acc00 = wmma_bf16(a0, b0, acc00);
        acc01 = wmma_bf16(a0, b1, acc01);
        acc10 = wmma_bf16(a1, b0, acc10);
        acc11 = wmma_bf16(a1, b1, acc11);
    }
#pragma unroll
    for (int v = 0; v < 8; ++v) {
        size_t r0 = (size_t)(mo + v + 8 * khalf) * Npad;
        size_t r1 = (size_t)(mo + 16 + v + 8 * khalf) * Npad;
        C[r0 + no + l15]      = acc00[v];
        C[r0 + no + 16 + l15] = acc01[v];
        C[r1 + no + l15]      = acc10[v];
        C[r1 + no + 16 + l15] = acc11[v];
    }
}

// ---- sequential RNN scan: one workgroup, W_hh resident in LDS (CDNA5 320KB/WGP) ----
// H[t] = tanh(X[t] + H[t-1] @ Whh^T), X precomputed = x@Wih^T + b_ih + b_hh
__global__ void __launch_bounds__(256) rnn_scan_kernel(
    const float* __restrict__ X,             // [S*B][HP] f32 (m = t*B + b)
    const unsigned short* __restrict__ Whh,  // [HP][HP] bf16
    unsigned short* __restrict__ Hseq) {     // [S*B][HP] bf16 out
    extern __shared__ unsigned short lds[];
    unsigned short* Wl    = lds;             // HP*HP      = 200 KB
    unsigned short* Hbuf0 = lds + HP * HP;   // BB*HP      =  20 KB
    unsigned short* Hbuf1 = Hbuf0 + BB * HP; //            =  20 KB
    for (int i = threadIdx.x; i < HP * HP / 8; i += 256)
        ((u32x4*)Wl)[i] = ((const u32x4*)Whh)[i];
    for (int i = threadIdx.x; i < BB * HP; i += 256) Hbuf0[i] = 0;
    __syncthreads();

    int lane = threadIdx.x & 31, wave = threadIdx.x >> 5;
    int khalf = lane >> 4, l15 = lane & 15;
    unsigned short* Hp = Hbuf0;
    unsigned short* Hn = Hbuf1;

    for (int t = 0; t < SS; ++t) {
        const float* Xt = X + (size_t)t * BB * HP;
        unsigned short* Hout = Hseq + (size_t)t * BB * HP;
        // 40 output tiles (2 M-tiles x 20 N-tiles), 5 per wave
#pragma unroll
        for (int ti = 0; ti < 5; ++ti) {
            int tile = wave * 5 + ti;
            int mo = (tile & 1) << 4;
            int no = (tile >> 1) << 4;
            v8f acc;
#pragma unroll
            for (int v = 0; v < 8; ++v)
                acc[v] = Xt[(mo + v + 8 * khalf) * HP + no + l15];
            const unsigned short* Arow = Hp + (mo + l15) * HP;
            const unsigned short* Brow = Wl + (no + l15) * HP;
#pragma unroll
            for (int kt = 0; kt < HP; kt += 32) {
                BF16Tile a, b;
                loadA(a, Arow, kt, khalf);
                loadB(b, Brow, kt, khalf);
                acc = wmma_bf16(a, b, acc);
            }
#pragma unroll
            for (int v = 0; v < 8; ++v) {
                float h = tanhf(acc[v]);
                unsigned short hb = f2bf(h);
                int off = (mo + v + 8 * khalf) * HP + no + l15;
                Hn[off]   = hb;
                Hout[off] = hb;
            }
        }
        __syncthreads();
        unsigned short* tmp = Hp; Hp = Hn; Hn = tmp;
    }
}

// ---- FC head, 2x2 blocked: out[b*S+s][NF] = H[s*B+b][:] @ Wf^T + bias ----
__global__ void __launch_bounds__(256) wmma_fc_kernel(
    const unsigned short* __restrict__ Hseq, const unsigned short* __restrict__ Wf,
    const float* __restrict__ bias, float* __restrict__ out) {
    int gw   = (int)((blockIdx.x * blockDim.x + threadIdx.x) >> 5);
    int lane = threadIdx.x & 31;
    int tilesN = NF >> 5;                      // 4
    int mo = (gw / tilesN) << 5;
    int no = (gw - (gw / tilesN) * tilesN) << 5;
    if (mo >= BB * SS) return;
    int khalf = lane >> 4, l15 = lane & 15;

    float bv0 = bias[no + l15];
    float bv1 = bias[no + 16 + l15];
    v8f acc00, acc01, acc10, acc11;
#pragma unroll
    for (int v = 0; v < 8; ++v) { acc00[v] = bv0; acc01[v] = bv1; acc10[v] = bv0; acc11[v] = bv1; }

    const unsigned short* Arow0 = Hseq + (size_t)(mo + l15) * HP;
    const unsigned short* Arow1 = Hseq + (size_t)(mo + 16 + l15) * HP;
    const unsigned short* Wcol0 = Wf + (size_t)(no + l15) * HP;
    const unsigned short* Wcol1 = Wf + (size_t)(no + 16 + l15) * HP;
#pragma unroll
    for (int kt = 0; kt < HP; kt += 32) {
        BF16Tile a0, a1, b0, b1;
        loadA(a0, Arow0, kt, khalf);
        loadA(a1, Arow1, kt, khalf);
        loadB(b0, Wcol0, kt, khalf);
        loadB(b1, Wcol1, kt, khalf);
        acc00 = wmma_bf16(a0, b0, acc00);
        acc01 = wmma_bf16(a0, b1, acc01);
        acc10 = wmma_bf16(a1, b0, acc10);
        acc11 = wmma_bf16(a1, b1, acc11);
    }
#pragma unroll
    for (int v = 0; v < 8; ++v) {
        int m0 = mo + v + 8 * khalf;
        int m1 = m0 + 16;
        size_t r0 = (size_t)((m0 & 31) * SS + (m0 >> 5)) * NF;
        size_t r1 = (size_t)((m1 & 31) * SS + (m1 >> 5)) * NF;
        out[r0 + no + l15]      = acc00[v];
        out[r0 + no + 16 + l15] = acc01[v];
        out[r1 + no + l15]      = acc10[v];
        out[r1 + no + 16 + l15] = acc11[v];
    }
}

extern "C" void kernel_launch(void* const* d_in, const int* in_sizes, int n_in,
                              void* d_out, int out_size, void* d_ws, size_t ws_size,
                              hipStream_t stream) {
    const int*   x    = (const int*)d_in[0];
    const float* emb  = (const float*)d_in[1];
    const float* Wih1 = (const float*)d_in[2];
    const float* Whh1 = (const float*)d_in[3];
    const float* bih1 = (const float*)d_in[4];
    const float* bhh1 = (const float*)d_in[5];
    const float* Wih2 = (const float*)d_in[6];
    const float* Whh2 = (const float*)d_in[7];
    const float* bih2 = (const float*)d_in[8];
    const float* bhh2 = (const float*)d_in[9];
    const float* fc1w = (const float*)d_in[10];
    const float* fc1b = (const float*)d_in[11];
    const float* fc2w = (const float*)d_in[12];
    const float* fc2b = (const float*)d_in[13];
    float* out = (float*)d_out;

    char* ws = (char*)d_ws;
    size_t o = 0;
    auto alloc = [&](size_t bytes) { size_t r = o; o += (bytes + 255) & ~(size_t)255; return r; };
    size_t oW1   = alloc((size_t)HP * K1P * 2);
    size_t oWhh1 = alloc((size_t)HP * HP * 2);
    size_t oWih2 = alloc((size_t)HP * HP * 2);
    size_t oWhh2 = alloc((size_t)HP * HP * 2);
    size_t oWf1  = alloc((size_t)NF * HP * 2);
    size_t oWf2  = alloc((size_t)NF * HP * 2);
    size_t oB1   = alloc(HP * 4);
    size_t oB2   = alloc(HP * 4);
    size_t oX1   = alloc((size_t)BB * SS * HP * 4);     // 21.0 MB
    size_t oBig  = alloc((size_t)BB * SS * K1P * 2);    // 49.3 MB, overlaid:
    size_t oXE = oBig;                                  //   XE   (dead after GEMM1)
    size_t oH1 = oBig;                                  //   H1seq 10.5 MB
    size_t oX2 = oBig + (size_t)BB * SS * HP * 2;       //   X2    21.0 MB
    size_t oH2 = oX2 + (size_t)BB * SS * HP * 4;        //   H2seq 10.5 MB

    unsigned short* W1b   = (unsigned short*)(ws + oW1);
    unsigned short* Whh1b = (unsigned short*)(ws + oWhh1);
    unsigned short* Wih2b = (unsigned short*)(ws + oWih2);
    unsigned short* Whh2b = (unsigned short*)(ws + oWhh2);
    unsigned short* Wf1b  = (unsigned short*)(ws + oWf1);
    unsigned short* Wf2b  = (unsigned short*)(ws + oWf2);
    float* B1 = (float*)(ws + oB1);
    float* B2 = (float*)(ws + oB2);
    float* X1 = (float*)(ws + oX1);
    unsigned short* XE    = (unsigned short*)(ws + oXE);
    unsigned short* H1seq = (unsigned short*)(ws + oH1);
    float*          X2    = (float*)(ws + oX2);
    unsigned short* H2seq = (unsigned short*)(ws + oH2);

    dim3 blk(256);
    // weight prep (bf16, zero-padded 300->320, 1500->1504)
    padcvt_kernel<<<(HP * K1P + 255) / 256, blk, 0, stream>>>(Wih1, W1b, HH, K1, HP, K1P);
    padcvt_kernel<<<(HP * HP + 255) / 256, blk, 0, stream>>>(Whh1, Whh1b, HH, HH, HP, HP);
    padcvt_kernel<<<(HP * HP + 255) / 256, blk, 0, stream>>>(Wih2, Wih2b, HH, HH, HP, HP);
    padcvt_kernel<<<(HP * HP + 255) / 256, blk, 0, stream>>>(Whh2, Whh2b, HH, HH, HP, HP);
    padcvt_kernel<<<(NF * HP + 255) / 256, blk, 0, stream>>>(fc1w, Wf1b, NF, HH, NF, HP);
    padcvt_kernel<<<(NF * HP + 255) / 256, blk, 0, stream>>>(fc2w, Wf2b, NF, HH, NF, HP);
    bias_kernel<<<(HP + 255) / 256, blk, 0, stream>>>(bih1, bhh1, B1, HH, HP);
    bias_kernel<<<(HP + 255) / 256, blk, 0, stream>>>(bih2, bhh2, B2, HH, HP);

    // embedding gather + relu -> bf16 [16384][1504]
    embed_kernel<<<(BB * SS * K1P + 255) / 256, blk, 0, stream>>>(x, emb, XE);

    const int M = BB * SS;                 // 16384
    // GEMM1 (bulk of FLOPs): X1 = XE @ Wih1^T + (b_ih1+b_hh1)   [2x2 blocked waves]
    int nW1 = (M / 32) * (HP / 32);        // 5120 waves
    wmma_gemm_kernel<<<nW1 / 8, blk, 0, stream>>>(XE, W1b, B1, X1, M, HP, K1P);

    // scan layer 1 (serial, single WGP, Whh1 in LDS)
    size_t ldsBytes = (size_t)(HP * HP + 2 * BB * HP) * 2;   // 240 KB
    rnn_scan_kernel<<<1, blk, ldsBytes, stream>>>(X1, Whh1b, H1seq);

    // GEMM2 (hoisted input term of layer 2): X2 = H1seq @ Wih2^T + (b_ih2+b_hh2)
    wmma_gemm_kernel<<<nW1 / 8, blk, 0, stream>>>(H1seq, Wih2b, B2, X2, M, HP, HP);

    // scan layer 2
    rnn_scan_kernel<<<1, blk, ldsBytes, stream>>>(X2, Whh2b, H2seq);

    // FC heads (permuted store seq-major -> batch-major)
    int nWF = (M / 32) * (NF / 32);        // 2048 waves
    wmma_fc_kernel<<<nWF / 8, blk, 0, stream>>>(H1seq, Wf1b, fc1b, out);
    wmma_fc_kernel<<<nWF / 8, blk, 0, stream>>>(H2seq, Wf2b, fc2b, out + (size_t)M * NF);
}
